// Net_33646773797613
// MI455X (gfx1250) — matrix operand order repaired and checked by reading, not compiled, not run on hardware
//
#include <hip/hip_runtime.h>
#include <hip/hip_bf16.h>
#include <stdint.h>

#define MM 4800          // BATCH*PER
#define ECHUNK 98304     // edge chunk (multiple of 16)

typedef __attribute__((ext_vector_type(16))) __bf16 v16bf;
typedef __attribute__((ext_vector_type(8)))  float  v8f;

union Frag { uint4 u[2]; v16bf v; };

__device__ __forceinline__ __bf16 f2bf(float f) { return (__bf16)f; }
__device__ __forceinline__ float sigm(float x)  { return 1.0f / (1.0f + __expf(-x)); }

// ---------------------------------------------------------------------------
// WMMA GEMM: C[rows,N] (f32) = A[rows,K](bf16, row-major, optional row gather)
//            x Bp (packed B fragments), optional bias / accumulate / relu.
// One wave computes a 32x64 output macro-tile: 2 M-tiles x 4 N-tiles = 8
// accumulators, 8 x v_wmma_f32_16x16x32_bf16 per 32-K chunk against
// 4 A-loads + 8 B-loads (b128).  N must be a multiple of 64.
// Packed B layout: tile (kt,nt) at ((kt*(N/16)+nt)*512) elements;
// lane l holds column nt*16+(l&15), K = kt*32 + (j<8 ? h*8+j : 16+h*8+j-8), h=(l>=16), j=0..15.
// ---------------------------------------------------------------------------
__global__ __launch_bounds__(32)
void gemm_bf16_wmma(const __bf16* __restrict__ A, const __bf16* __restrict__ Bp,
                    float* __restrict__ C, const float* __restrict__ bias,
                    const int* __restrict__ rowidx,
                    int rows, int K, int N, int accFlag, int reluFlag)
{
    const int NT   = N >> 4;          // 16-wide tiles in N
    const int NT4  = N >> 6;          // 64-wide macro-tiles in N
    const int tile = blockIdx.x;
    const int mt   = tile / NT4;      // 32-row macro-tile index
    const int nt4  = tile - mt * NT4;
    const int ntB  = nt4 * 4;         // first 16-wide N tile
    const int l    = threadIdx.x;     // 0..31
    const int half = (l >> 4) & 1;

    int r0 = mt * 32 + (l & 15);
    int r1 = r0 + 16;
    if (r0 >= rows) r0 = rows - 1;
    if (r1 >= rows) r1 = rows - 1;
    const int ar0 = rowidx ? rowidx[r0] : r0;
    const int ar1 = rowidx ? rowidx[r1] : r1;

    const __bf16* ap0 = A + (size_t)ar0 * K + half * 8;
    const __bf16* ap1 = A + (size_t)ar1 * K + half * 8;

    v8f acc[2][4];
#pragma unroll
    for (int i = 0; i < 2; ++i)
#pragma unroll
        for (int j = 0; j < 4; ++j) acc[i][j] = (v8f){};

    const int KT = K >> 5;
    for (int kt = 0; kt < KT; ++kt) {
        Frag fa0, fa1, fb[4];
        const __bf16* a0 = ap0 + kt * 32;
        const __bf16* a1 = ap1 + kt * 32;
        fa0.u[0] = *(const uint4*)(a0);
        fa0.u[1] = *(const uint4*)(a0 + 16);
        fa1.u[0] = *(const uint4*)(a1);
        fa1.u[1] = *(const uint4*)(a1 + 16);
        const __bf16* bbase = Bp + ((size_t)kt * NT + ntB) * 512 + (size_t)l * 16;
#pragma unroll
        for (int j = 0; j < 4; ++j) {
            const __bf16* b0 = bbase + j * 512;
            fb[j].u[0] = *(const uint4*)(b0);
            fb[j].u[1] = *(const uint4*)(b0 + 8);
        }
        if (kt + 1 < KT)
            __builtin_prefetch(Bp + ((size_t)(kt + 1) * NT + ntB) * 512 + (size_t)l * 16, 0, 3);
#pragma unroll
        for (int j = 0; j < 4; ++j) {
            acc[0][j] = __builtin_amdgcn_wmma_f32_16x16x32_bf16(
                            false, fa0.v, false, fb[j].v, (short)0, acc[0][j], false, false);
            acc[1][j] = __builtin_amdgcn_wmma_f32_16x16x32_bf16(
                            false, fa1.v, false, fb[j].v, (short)0, acc[1][j], false, false);
        }
    }

#pragma unroll
    for (int j = 0; j < 4; ++j) {
        const int   n  = (ntB + j) * 16 + (l & 15);
        const float bn = bias ? bias[n] : 0.0f;
#pragma unroll
        for (int i = 0; i < 2; ++i) {
#pragma unroll
            for (int v = 0; v < 8; ++v) {
                int m = mt * 32 + i * 16 + v + half * 8;
                if (m < rows) {
                    size_t offc = (size_t)m * N + n;
                    float val = acc[i][j][v] + bn;
                    if (accFlag)  val += C[offc];
                    if (reluFlag) val = fmaxf(val, 0.0f);
                    C[offc] = val;
                }
            }
        }
    }
}

// Pack a KxN f32 matrix (element [k][n] = src[k*sK + n*sN]) into the
// bf16 WMMA B-fragment layout described above.
__global__ void pack_b_bf16(__bf16* __restrict__ dst, const float* __restrict__ src,
                            int K, int N, int sK, int sN)
{
    int t = blockIdx.x * blockDim.x + threadIdx.x;
    int total = K * N;
    if (t >= total) return;
    int NT   = N >> 4;
    int tile = t >> 9;
    int e    = t & 511;
    int lane = e >> 4;
    int j    = e & 15;
    int kt   = tile / NT;
    int nt   = tile - kt * NT;
    int k    = kt * 32 + ((lane >= 16) ? 8 : 0) + ((j < 8) ? j : 16 + (j - 8));
    int n    = nt * 16 + (lane & 15);
    dst[t]   = f2bf(src[(size_t)k * sK + (size_t)n * sN]);
}

// ---------------------------------------------------------------------------
// Fused conv1d (k=1 or 3) + GroupNorm over (C,L) per sample + optional
// residual add + optional relu.  One block (256 threads) per sample.
// ---------------------------------------------------------------------------
__global__ __launch_bounds__(256)
void conv_gn(const float* __restrict__ in, const float* __restrict__ w,
             const float* __restrict__ gam, const float* __restrict__ bet,
             const float* __restrict__ resid, float* __restrict__ out,
             int Cin, int Cout, int Lin, int Lout,
             int stride, int pad, int ks, int doRelu)
{
    __shared__ float sbuf[2560];
    __shared__ float rs[256];
    __shared__ float rq[256];
    const int m   = blockIdx.x;
    const int tid = threadIdx.x;
    const int CL  = Cout * Lout;
    const float* xin = in + (size_t)m * Cin * Lin;

    float psum = 0.f, psq = 0.f;
    for (int idx = tid; idx < CL; idx += blockDim.x) {
        int c = idx / Lout;
        int x = idx - c * Lout;
        float a = 0.f;
        const float* wc = w + (size_t)c * Cin * ks;
        for (int ci = 0; ci < Cin; ++ci) {
            const float* xr = xin + ci * Lin;
            const float* wr = wc + ci * ks;
            for (int t = 0; t < ks; ++t) {
                int xi = x * stride - pad + t;
                if (xi >= 0 && xi < Lin) a += xr[xi] * wr[t];
            }
        }
        sbuf[idx] = a;
        psum += a; psq += a * a;
    }
    rs[tid] = psum; rq[tid] = psq;
    __syncthreads();
    for (int s = blockDim.x >> 1; s > 0; s >>= 1) {
        if (tid < s) { rs[tid] += rs[tid + s]; rq[tid] += rq[tid + s]; }
        __syncthreads();
    }
    const float invn = 1.0f / (float)CL;
    const float mean = rs[0] * invn;
    const float var  = rq[0] * invn - mean * mean;
    const float rstd = rsqrtf(var + 1e-5f);
    for (int idx = tid; idx < CL; idx += blockDim.x) {
        int c = idx / Lout;
        float v = (sbuf[idx] - mean) * rstd * gam[c] + bet[c];
        if (resid)  v += resid[(size_t)m * CL + idx];
        if (doRelu) v = fmaxf(v, 0.f);
        out[(size_t)m * CL + idx] = v;
    }
}

// out[t] = linear_upsample2(x)[t] + lat[t]    (x: (rows*C, Lin) -> 2*Lin)
__global__ void upsample_add(const float* __restrict__ x, const float* __restrict__ lat,
                             float* __restrict__ out, int Lin, int total)
{
    int t = blockIdx.x * blockDim.x + threadIdx.x;
    if (t >= total) return;
    int Lo = 2 * Lin;
    int i  = t % Lo;
    int mc = t / Lo;
    float pos  = fmaxf((i + 0.5f) * 0.5f - 0.5f, 0.0f);
    int   lo   = (int)floorf(pos);
    float frac = pos - (float)lo;
    int   hi2  = lo + 1; if (hi2 > Lin - 1) hi2 = Lin - 1;
    const float* xr = x + (size_t)mc * Lin;
    out[t] = xr[lo] * (1.0f - frac) + xr[hi2] * frac + lat[t];
}

// LayerNorm over 128 features per row, per-feature affine, optional relu.
// Writes f32 and/or bf16. One 128-thread block per row.
__global__ __launch_bounds__(128)
void rowgn128(const float* __restrict__ in, float* __restrict__ outf,
              __bf16* __restrict__ outb, const float* __restrict__ gam,
              const float* __restrict__ bet, int doRelu)
{
    __shared__ float rs[128];
    __shared__ float rq[128];
    int row = blockIdx.x;
    int j   = threadIdx.x;
    float v = in[(size_t)row * 128 + j];
    rs[j] = v; rq[j] = v * v;
    __syncthreads();
    for (int s = 64; s > 0; s >>= 1) {
        if (j < s) { rs[j] += rs[j + s]; rq[j] += rq[j + s]; }
        __syncthreads();
    }
    float mean = rs[0] * (1.0f / 128.0f);
    float var  = rq[0] * (1.0f / 128.0f) - mean * mean;
    float rstd = rsqrtf(var + 1e-5f);
    float o = (v - mean) * rstd * gam[j] + bet[j];
    if (doRelu) o = fmaxf(o, 0.f);
    if (outf) outf[(size_t)row * 128 + j] = o;
    if (outb) outb[(size_t)row * 128 + j] = f2bf(o);
}

// d = relu((ctrs[hi]-ctrs[wi]) @ dw0 + db0)  -> bf16
__global__ void edge_d0(const float* __restrict__ ctrs, const int* __restrict__ hi,
                        const int* __restrict__ wi, const float* __restrict__ dw0,
                        const float* __restrict__ db0, __bf16* __restrict__ out, int E)
{
    int t = blockIdx.x * blockDim.x + threadIdx.x;
    if (t >= E * 128) return;
    int e = t >> 7, j = t & 127;
    float x0 = ctrs[hi[e] * 2 + 0] - ctrs[wi[e] * 2 + 0];
    float x1 = ctrs[hi[e] * 2 + 1] - ctrs[wi[e] * 2 + 1];
    float v  = fmaxf(x0 * dw0[j] + x1 * dw0[128 + j] + db0[j], 0.f);
    out[t] = f2bf(v);
}

__global__ void scatter_add128(float* __restrict__ a, const float* __restrict__ c,
                               const int* __restrict__ hi, int E)
{
    int t = blockIdx.x * blockDim.x + threadIdx.x;
    if (t >= E * 128) return;
    int e = t >> 7, j = t & 127;
    atomicAdd(&a[(size_t)hi[e] * 128 + j], c[t]);
}

// resf = relu(x + resf); outb = bf16(resf)
__global__ void add_relu_bf(const float* __restrict__ x, float* __restrict__ resf,
                            __bf16* __restrict__ outb, int total)
{
    int t = blockIdx.x * blockDim.x + threadIdx.x;
    if (t >= total) return;
    float v = fmaxf(x[t] + resf[t], 0.f);
    resf[t] = v; outb[t] = f2bf(v);
}

__global__ void vec_add(const float* a, const float* b, float* o, int n)
{
    int t = blockIdx.x * blockDim.x + threadIdx.x;
    if (t < n) o[t] = a[t] + b[t];
}

// h = c = ctrs @ h0_w  ((rows,2)@(2,128))
__global__ void h0_init(const float* __restrict__ ctrs, const float* __restrict__ W,
                        float* __restrict__ h, float* __restrict__ c, int rows)
{
    int t = blockIdx.x * blockDim.x + threadIdx.x;
    if (t >= rows * 128) return;
    int m = t >> 7, j = t & 127;
    float v = ctrs[m * 2] * W[j] + ctrs[m * 2 + 1] * W[128 + j];
    h[t] = v; c[t] = v;
}

// A[m, 0:128] = outb[m, :, t]  (seq features);  A[m, 128:256] = h[m, :]
__global__ void pack_xh(const float* __restrict__ outb, const float* __restrict__ h,
                        __bf16* __restrict__ A, int tstep, int rows)
{
    int t = blockIdx.x * blockDim.x + threadIdx.x;
    if (t >= rows * 256) return;
    int m = t >> 8, k = t & 255;
    float v = (k < 128) ? outb[(size_t)m * 2560 + k * 20 + tstep]
                        : h[(size_t)m * 128 + (k - 128)];
    A[t] = f2bf(v);
}

__global__ void lstm_gates(const float* __restrict__ G, float* __restrict__ h,
                           float* __restrict__ c, int rows)
{
    int t = blockIdx.x * blockDim.x + threadIdx.x;
    if (t >= rows * 128) return;
    int m = t >> 7, j = t & 127;
    const float* g = G + (size_t)m * 512;
    float ig = sigm(g[j]);
    float fg = sigm(g[128 + j]);
    float gg = tanhf(g[256 + j]);
    float og = sigm(g[384 + j]);
    float cn = fg * c[t] + ig * gg;
    c[t] = cn;
    h[t] = og * tanhf(cn);
}

// agts = h + out[:, :, 19]
__global__ void add_last(const float* __restrict__ h, const float* __restrict__ outb,
                         float* __restrict__ agf, __bf16* __restrict__ agb, int rows)
{
    int t = blockIdx.x * blockDim.x + threadIdx.x;
    if (t >= rows * 128) return;
    int m = t >> 7, j = t & 127;
    float v = h[t] + outb[(size_t)m * 2560 + j * 20 + 19];
    agf[t] = v; agb[t] = f2bf(v);
}

// ---------------------------------------------------------------------------
// Host orchestration
// ---------------------------------------------------------------------------
extern "C" void kernel_launch(void* const* d_in, const int* in_sizes, int n_in,
                              void* d_out, int out_size, void* d_ws, size_t ws_size,
                              hipStream_t stream)
{
    (void)n_in; (void)out_size; (void)ws_size;
    auto P = [&](int i) -> const float* { return (const float*)d_in[i]; };

    // Input order: actors, actor_ctrs, then params pytree leaves (dict keys
    // sorted, lists in order), then hi, wi.
    const float* actors = P(0);
    const float* ctrs   = P(1);
    const int*   hi     = (const int*)d_in[103];
    const int*   wi     = (const int*)d_in[104];
    const int    E      = in_sizes[103];

    // ---- workspace allocator ----
    char*  base = (char*)d_ws;
    size_t off  = 0;
    auto alloc = [&](size_t bytes) -> void* {
        void* p = base + off;
        off = (off + bytes + 255) & ~(size_t)255;
        return p;
    };

    float* T1   = (float*)alloc((size_t)MM * 128 * 20 * 4);
    float* T2   = (float*)alloc((size_t)MM * 128 * 20 * 4);
    float* OUTB = (float*)alloc((size_t)MM * 128 * 20 * 4);
    float* F0   = (float*)alloc((size_t)MM * 16 * 20 * 4);
    float* F1   = (float*)alloc((size_t)MM * 32 * 10 * 4);
    float* F2   = (float*)alloc((size_t)MM * 64 * 5 * 4);
    float* hb   = (float*)alloc((size_t)MM * 128 * 4);
    float* cb   = (float*)alloc((size_t)MM * 128 * 4);
    __bf16* Acat = (__bf16*)alloc((size_t)MM * 256 * 2);
    float* G    = (float*)alloc((size_t)MM * 512 * 4);
    float* biasS= (float*)alloc(512 * 4);
    float* agf  = (float*)alloc((size_t)MM * 128 * 4);
    __bf16* agb = (__bf16*)alloc((size_t)MM * 128 * 2);
    float* aM   = (float*)alloc((size_t)MM * 128 * 4);
    float* aT   = (float*)alloc((size_t)MM * 128 * 4);
    __bf16* aBF = (__bf16*)alloc((size_t)MM * 128 * 2);
    __bf16* WcatP = (__bf16*)alloc((size_t)256 * 512 * 2);
    __bf16* ebD = (__bf16*)alloc((size_t)ECHUNK * 128 * 2);
    __bf16* ebQ = (__bf16*)alloc((size_t)ECHUNK * 128 * 2);
    __bf16* ebC = (__bf16*)alloc((size_t)ECHUNK * 128 * 2);
    float* ef32 = (float*)alloc((size_t)ECHUNK * 128 * 4);

    auto gemm = [&](const __bf16* A, const __bf16* Bp, float* C, const float* bias,
                    const int* ridx, int rows, int K, int N, int acc, int relu) {
        int mt = (rows + 31) >> 5, nt4 = N >> 6;
        gemm_bf16_wmma<<<dim3(mt * nt4), dim3(32), 0, stream>>>(
            A, Bp, C, bias, ridx, rows, K, N, acc, relu);
    };
    auto packW = [&](__bf16* dst, const float* src, int K, int N, int sK, int sN) {
        int total = K * N;
        pack_b_bf16<<<(total + 255) / 256, 256, 0, stream>>>(dst, src, K, N, sK, sN);
    };
    auto convgn = [&](const float* x, int wI, int gI, int bI, const float* resid,
                      float* out, int Cin, int Cout, int Lin, int Lout,
                      int stride, int pad, int ks, int relu) {
        conv_gn<<<MM, 256, 0, stream>>>(x, P(wI), P(gI), P(bI), resid, out,
                                        Cin, Cout, Lin, Lout, stride, pad, ks, relu);
    };
    // res1d: params at base in sorted-key order
    //   with wd:    b1 b2 bd g1 g2 gd w1 w2 wd  (base+0..8)
    //   without wd: b1 b2 g1 g2 w1 w2           (base+0..5)
    auto res1d = [&](const float* x, float* out, float* t1, float* t2,
                     int b, bool wd, int Cin, int Cout, int Lin, int Lout, int stride) {
        if (wd) {
            convgn(x,  b + 6, b + 3, b + 0, nullptr, t1, Cin, Cout, Lin, Lout, stride, 1, 3, 1);
            convgn(x,  b + 8, b + 5, b + 2, nullptr, t2, Cin, Cout, Lin, Lout, stride, 0, 1, 0);
            convgn(t1, b + 7, b + 4, b + 1, t2, out, Cout, Cout, Lout, Lout, 1, 1, 3, 1);
        } else {
            convgn(x,  b + 4, b + 2, b + 0, nullptr, t1, Cin, Cout, Lin, Lout, stride, 1, 3, 1);
            convgn(t1, b + 5, b + 3, b + 1, x, out, Cout, Cout, Lout, Lout, 1, 1, 3, 1);
        }
    };

    // ---- pack GEMM weights (bf16 B-fragments) ----
    packW(WcatP,          P(60), 128, 512, 1, 128);   // wih^T  (K 0..127)
    packW(WcatP + 65536,  P(59), 128, 512, 1, 128);   // whh^T  (K 128..255)
    __bf16* pk[2][8];
    for (int a = 0; a < 2; ++a) {
        int b = 67 + a * 18; // att leaves: aw cb0 cg0 cw0 cw1 db0 db1 dg1 dw0 dw1 lb lg lw nb ng qb qg qw
        for (int i = 0; i < 8; ++i) pk[a][i] = (__bf16*)alloc((size_t)128 * 128 * 2);
        packW(pk[a][0], P(b + 9),           128, 128, 128, 1);  // dw1
        packW(pk[a][1], P(b + 17),          128, 128, 128, 1);  // qw
        packW(pk[a][2], P(b + 3) + 0*16384, 128, 128, 128, 1);  // cw0[0:128]
        packW(pk[a][3], P(b + 3) + 1*16384, 128, 128, 128, 1);  // cw0[128:256]
        packW(pk[a][4], P(b + 3) + 2*16384, 128, 128, 128, 1);  // cw0[256:384]
        packW(pk[a][5], P(b + 4),           128, 128, 128, 1);  // cw1
        packW(pk[a][6], P(b + 0),           128, 128, 128, 1);  // aw
        packW(pk[a][7], P(b + 12),          128, 128, 128, 1);  // lw
    }

    // ---- actor net front-end ----
    res1d(actors, F0, T1, T2,  2, true,   3, 16, 20, 20, 1);
    res1d(F0,     F0, T1, T2, 11, false, 16, 16, 20, 20, 1);
    res1d(F0,     F1, T1, T2, 17, true,  16, 32, 20, 10, 2);
    res1d(F1,     F1, T1, T2, 26, false, 32, 32, 10, 10, 1);
    res1d(F1,     F2, T1, T2, 32, true,  32, 64, 10,  5, 2);
    res1d(F2,     F2, T1, T2, 41, false, 64, 64,  5,  5, 1);

    // FPN: lateral convs + upsample-add (lat dicts: b,g,w at 48/51/54)
    convgn(F2, 56, 55, 54, nullptr, T1, 64, 128, 5, 5, 1, 1, 3, 0);
    convgn(F1, 53, 52, 51, nullptr, T2, 32, 128, 10, 10, 1, 1, 3, 0);
    upsample_add<<<(MM * 128 * 10 + 255) / 256, 256, 0, stream>>>(T1, T2, T2, 5,  MM * 128 * 10);
    convgn(F0, 50, 49, 48, nullptr, T1, 16, 128, 20, 20, 1, 1, 3, 0);
    upsample_add<<<(MM * 128 * 20 + 255) / 256, 256, 0, stream>>>(T2, T1, T1, 10, MM * 128 * 20);

    // out_block (128->128, no wd) at 61
    res1d(T1, OUTB, T2, T2, 61, false, 128, 128, 20, 20, 1);

    // ---- LSTM (20 dependent steps; each step one WMMA GEMM K=256,N=512) ----
    h0_init<<<(MM * 128 + 255) / 256, 256, 0, stream>>>(ctrs, P(47), hb, cb, MM);
    vec_add<<<2, 256, 0, stream>>>(P(58), P(57), biasS, 512);   // bih + bhh
    for (int t = 0; t < 20; ++t) {
        pack_xh<<<(MM * 256 + 255) / 256, 256, 0, stream>>>(OUTB, hb, Acat, t, MM);
        gemm(Acat, WcatP, G, biasS, nullptr, MM, 256, 512, 0, 0);
        lstm_gates<<<(MM * 128 + 255) / 256, 256, 0, stream>>>(G, hb, cb, MM);
    }
    add_last<<<(MM * 128 + 255) / 256, 256, 0, stream>>>(hb, OUTB, agf, agb, MM);

    // ---- two attention layers ----
    for (int a = 0; a < 2; ++a) {
        int b = 67 + a * 18;
        const float *dw0 = P(b + 8), *db0 = P(b + 5);
        const float *dg1 = P(b + 7), *db1 = P(b + 6);
        const float *qg  = P(b + 16), *qb = P(b + 15);
        const float *cg0 = P(b + 2),  *cb0 = P(b + 1);
        const float *ng  = P(b + 14), *nb = P(b + 13);
        const float *lg  = P(b + 11), *lb = P(b + 10);

        gemm(agb, pk[a][6], aM, nullptr, nullptr, MM, 128, 128, 0, 0);   // a = agts @ aw

        for (int e0 = 0; e0 < E; e0 += ECHUNK) {
            int ec = (E - e0 < ECHUNK) ? (E - e0) : ECHUNK;
            int gE = (ec * 128 + 255) / 256;
            edge_d0<<<gE, 256, 0, stream>>>(ctrs, hi + e0, wi + e0, dw0, db0, ebC, ec);
            gemm(ebC, pk[a][0], ef32, nullptr, nullptr, ec, 128, 128, 0, 0);     // d @ dw1
            rowgn128<<<ec, 128, 0, stream>>>(ef32, nullptr, ebD, dg1, db1, 1);
            gemm(agb, pk[a][1], ef32, nullptr, hi + e0, ec, 128, 128, 0, 0);     // agts[hi] @ qw
            rowgn128<<<ec, 128, 0, stream>>>(ef32, nullptr, ebQ, qg, qb, 1);
            gemm(ebD, pk[a][2], ef32, nullptr, nullptr,  ec, 128, 128, 0, 0);    // d @ cw0a
            gemm(ebQ, pk[a][3], ef32, nullptr, nullptr,  ec, 128, 128, 1, 0);    // + q @ cw0b
            gemm(agb, pk[a][4], ef32, nullptr, wi + e0,  ec, 128, 128, 1, 0);    // + agts[wi] @ cw0c
            rowgn128<<<ec, 128, 0, stream>>>(ef32, nullptr, ebC, cg0, cb0, 1);
            gemm(ebC, pk[a][5], ef32, nullptr, nullptr,  ec, 128, 128, 0, 0);    // c @ cw1
            scatter_add128<<<gE, 256, 0, stream>>>(aM, ef32, hi + e0, ec);
        }
        rowgn128<<<MM, 128, 0, stream>>>(aM, aM, aBF, ng, nb, 1);
        gemm(aBF, pk[a][7], aT, nullptr, nullptr, MM, 128, 128, 0, 0);           // @ lw
        rowgn128<<<MM, 128, 0, stream>>>(aT, aT, nullptr, lg, lb, 0);
        add_relu_bf<<<(MM * 128 + 255) / 256, 256, 0, stream>>>(aT, agf, agb, MM * 128);
    }

    hipMemcpyAsync(d_out, agf, (size_t)MM * 128 * 4, hipMemcpyDeviceToDevice, stream);
}
